// SelfAttention_50027779064199
// MI455X (gfx1250) — matrix operand order repaired and checked by reading, not compiled, not run on hardware
//
#include <hip/hip_runtime.h>

#define DEVINL __device__ __forceinline__

typedef __attribute__((ext_vector_type(2)))  __bf16 v2bf;
typedef __attribute__((ext_vector_type(8)))  __bf16 v8bf;
typedef __attribute__((ext_vector_type(16))) __bf16 v16bf;
typedef __attribute__((ext_vector_type(8)))  float  v8f;

// Problem constants (match reference)
constexpr int BATCH = 4;
constexpr int C     = 512;
constexpr int CK    = 64;           // low-rank q/k dim
constexpr int NPIX  = 4096;         // H*W
constexpr int ROWS  = 2 * CK + C;   // 640 fused projection rows

// Workspace (bf16 elements): fT [B][N][CK] | gT [B][N][CK] | hv [B][C][N]
constexpr size_t FG_ELEMS = (size_t)BATCH * NPIX * CK;

// ---------- helpers ----------

DEVINL unsigned short f2bf(float f) {
  __bf16 h = (__bf16)f;
  return __builtin_bit_cast(unsigned short, h);
}

DEVINL unsigned int pack2(float a, float b) {   // v_cvt_pk_bf16_f32
  v2bf v; v.x = (__bf16)a; v.y = (__bf16)b;
  return __builtin_bit_cast(unsigned int, v);
}

DEVINL v16bf cat8(v8bf lo, v8bf hi) {
  return __builtin_shufflevector(lo, hi, 0,1,2,3,4,5,6,7,8,9,10,11,12,13,14,15);
}

// A-matrix fragment (16x32 bf16), row-major source, element stride.
DEVINL v16bf fragA(const unsigned short* base, int stride, int row0, int k0, int lane) {
  const int r  = row0 + (lane & 15);
  const int kb = k0 + ((lane & 16) ? 8 : 0);
  const unsigned short* p = base + r * stride + kb;
  v8bf lo = *reinterpret_cast<const v8bf*>(p);
  v8bf hi = *reinterpret_cast<const v8bf*>(p + 16);
  return cat8(lo, hi);
}

// B-matrix fragment (32x16 bf16) from TRANSPOSED storage BT[col][k].
DEVINL v16bf fragBT(const unsigned short* base, int stride, int col0, int k0, int lane) {
  const int cc = col0 + (lane & 15);
  const int kb = k0 + ((lane & 16) ? 8 : 0);
  const unsigned short* p = base + cc * stride + kb;
  v8bf lo = *reinterpret_cast<const v8bf*>(p);
  v8bf hi = *reinterpret_cast<const v8bf*>(p + 16);
  return cat8(lo, hi);
}

DEVINL v8f wmma_bf16(v16bf a, v16bf b, v8f c) {
  return __builtin_amdgcn_wmma_f32_16x16x32_bf16(false, a, false, b, (short)0, c, false, false);
}

DEVINL uint4 packD(const v8f& a) {
  uint4 q;
  q.x = pack2(a[0], a[1]); q.y = pack2(a[2], a[3]);
  q.z = pack2(a[4], a[5]); q.w = pack2(a[6], a[7]);
  return q;
}

DEVINL float max8(const v8f& a) {
  float m = fmaxf(fmaxf(fmaxf(a[0], a[1]), fmaxf(a[2], a[3])),
                  fmaxf(fmaxf(a[4], a[5]), fmaxf(a[6], a[7])));
  return m;
}

// =====================================================================
// Kernel 1: fused projection GEMM (f,g stored transposed; hv row-major).
// Grid: (N/64, 10, B), block 256 (8 waves, 4x2 tiling of a 64x64 tile).
// =====================================================================
__global__ __launch_bounds__(256) void proj_kernel(
    const float* __restrict__ x,  const float* __restrict__ Wq,
    const float* __restrict__ Wk, const float* __restrict__ Wv,
    unsigned short* __restrict__ fgh)
{
  constexpr int AS = 40;            // padded bf16 stride
  constexpr int STS = 72;           // store-bounce stride
  __shared__ __align__(16) unsigned short smem[64 * AS * 2];   // 10240 B
  unsigned short* As  = smem;            // W tile   [64 r][32 k]
  unsigned short* BsT = smem + 64 * AS;  // x tile^T [64 n][32 k]
  // overlay for the hv store bounce: [64][72] (9216 B <= 10240 B)

  const int t = threadIdx.x, lane = t & 31, wave = t >> 5;
  const int n0 = blockIdx.x * 64;
  const int mt = blockIdx.y;
  const int b  = blockIdx.z;

  const float* Wsel; int wrow0;
  if (mt == 0)      { Wsel = Wq; wrow0 = 0; }
  else if (mt == 1) { Wsel = Wk; wrow0 = 0; }
  else              { Wsel = Wv; wrow0 = (mt - 2) * 64; }

  const float* xb = x + (size_t)b * C * NPIX;

  const int rg = (wave >> 1) * 16;
  const int cg = (wave & 1) * 32;

  const v8f vz = {0.f,0.f,0.f,0.f,0.f,0.f,0.f,0.f};
  v8f acc0 = vz, acc1 = vz;

  for (int kt = 0; kt < C / 32; ++kt) {
    const int k0 = kt * 32;
    {   // stage W tile (fp32 -> bf16)
      const int row = t >> 2;
      const int kq  = (t & 3) * 8;
      const float* wp = Wsel + (size_t)(wrow0 + row) * C + k0 + kq;
      float4 f0 = *reinterpret_cast<const float4*>(wp);
      float4 f1 = *reinterpret_cast<const float4*>(wp + 4);
      uint4 q;
      q.x = pack2(f0.x, f0.y); q.y = pack2(f0.z, f0.w);
      q.z = pack2(f1.x, f1.y); q.w = pack2(f1.z, f1.w);
      *reinterpret_cast<uint4*>(&As[row * AS + kq]) = q;
    }
    {   // stage x tile transposed: BsT[n][k] = x[k0+k][n0+n]
      const int n  = t & 63;
      const int kb = (t >> 6) * 8;
      float v[8];
      #pragma unroll
      for (int i = 0; i < 8; ++i)
        v[i] = xb[(size_t)(k0 + kb + i) * NPIX + n0 + n];
      uint4 q;
      q.x = pack2(v[0], v[1]); q.y = pack2(v[2], v[3]);
      q.z = pack2(v[4], v[5]); q.w = pack2(v[6], v[7]);
      *reinterpret_cast<uint4*>(&BsT[n * AS + kb]) = q;
    }
    __syncthreads();

    v16bf a  = fragA (As,  AS, rg, 0, lane);
    v16bf b0 = fragBT(BsT, AS, cg,      0, lane);
    v16bf b1 = fragBT(BsT, AS, cg + 16, 0, lane);
    acc0 = wmma_bf16(a, b0, acc0);
    acc1 = wmma_bf16(a, b1, acc1);
    __syncthreads();
  }

  if (mt < 2) {
    // f / g: transposed [n][q]; 8 accumulator values are contiguous in q
    unsigned short* tbase = fgh + (mt ? FG_ELEMS : 0) + (size_t)b * NPIX * CK;
    const int r0   = rg + ((lane & 16) ? 8 : 0);
    const int ncol = n0 + cg + (lane & 15);
    *reinterpret_cast<uint4*>(&tbase[(size_t)ncol        * CK + r0]) = packD(acc0);
    *reinterpret_cast<uint4*>(&tbase[(size_t)(ncol + 16) * CK + r0]) = packD(acc1);
  } else {
    // hv: bounce through LDS for coalesced b128 stores
    unsigned short* St = smem;   // [64][STS], safe after loop-end barrier
    const int r0l = rg + ((lane & 16) ? 8 : 0);
    const int ncl = cg + (lane & 15);
    #pragma unroll
    for (int v = 0; v < 8; ++v) {
      St[(r0l + v) * STS + ncl]      = f2bf(acc0[v]);
      St[(r0l + v) * STS + ncl + 16] = f2bf(acc1[v]);
    }
    __syncthreads();
    unsigned short* hbase = fgh + 2 * FG_ELEMS + (size_t)b * C * NPIX;
    const int row = t >> 2, seg = (t & 3) * 16;
    uint4 a0 = *reinterpret_cast<const uint4*>(&St[row * STS + seg]);
    uint4 a1 = *reinterpret_cast<const uint4*>(&St[row * STS + seg + 8]);
    unsigned short* dst = &hbase[(size_t)((mt - 2) * 64 + row) * NPIX + n0 + seg];
    *reinterpret_cast<uint4*>(dst)     = a0;
    *reinterpret_cast<uint4*>(dst + 8) = a1;
  }
}

// =====================================================================
// Kernel 2: flash attention over n, computing O^T[m,c] so that D-fragments
// are m-contiguous per lane (vectorized finalize). Softmax is done in
// registers (shfl across half-waves); only P (bf16) lives in LDS.
// Grid: (N/64, B), block 256 (8 waves; wave w owns c-slice [64w,64w+64)).
// =====================================================================
__global__ __launch_bounds__(256) void attn_kernel(
    const float* __restrict__ x, const unsigned short* __restrict__ fgh,
    const float* __restrict__ gamma, float* __restrict__ out)
{
  constexpr int TS = 72;   // bf16 stride (144B) for PT
  __shared__ __align__(16) unsigned short PT[64 * TS];  // PT[m][n] bf16
  __shared__ __align__(16) float scld[64];
  __shared__ float red1[4][64], red2[4][64];
  __shared__ float Mld[64], Lld[64];

  const int t = threadIdx.x, lane = t & 31, wave = t >> 5;
  const int m0 = blockIdx.x * 64;
  const int b  = blockIdx.y;

  const unsigned short* fTg = fgh + (size_t)b * NPIX * CK;             // [n][q]
  const unsigned short* gTg = fgh + FG_ELEMS + (size_t)b * NPIX * CK;  // [m][q]
  const unsigned short* hb  = fgh + 2 * FG_ELEMS + (size_t)b * C * NPIX;

  if (t < 64) { Mld[t] = -3.0e38f; Lld[t] = 0.f; }
  __syncthreads();

  // Wave's S sub-tile: 16 n-rows x 32 m-cols (4x2 wave grid covers 64x64)
  const int sn0 = (wave >> 1) * 16;
  const int sm0 = (wave & 1) * 32;

  // Resident Q fragments straight from global gT (L2-resident)
  const unsigned short* gTm = gTg + (size_t)m0 * CK;
  v16bf Bq[2][2];
  #pragma unroll
  for (int mc = 0; mc < 2; ++mc)
    #pragma unroll
    for (int kk = 0; kk < 2; ++kk)
      Bq[mc][kk] = fragBT(gTm, CK, sm0 + mc * 16, kk * 32, lane);

  const v8f vz = {0.f,0.f,0.f,0.f,0.f,0.f,0.f,0.f};
  v8f accO[4][4];                 // [mf][cf] : O^T tile 64m x 64c
  #pragma unroll
  for (int mf = 0; mf < 4; ++mf)
    #pragma unroll
    for (int cf = 0; cf < 4; ++cf) accO[mf][cf] = vz;

  const int cbase = wave * 64;
  const int col   = lane & 15;
  const int hsel  = (lane & 16) ? 8 : 0;

  for (int nb = 0; nb < NPIX / 64; ++nb) {
    const int nblk = nb * 64;
    const unsigned short* fTn = fTg + (size_t)nblk * CK;
    if (nb + 1 < NPIX / 64)
      __builtin_prefetch(fTn + 64 * CK + (size_t)t * 16, 0, 1);

    // ---- S = f^T g (A-frags straight from global) ----
    v8f aS0 = vz, aS1 = vz;
    #pragma unroll
    for (int kk = 0; kk < 2; ++kk) {
      v16bf aF = fragA(fTn, CK, sn0, kk * 32, lane);
      aS0 = wmma_bf16(aF, Bq[0][kk], aS0);
      aS1 = wmma_bf16(aF, Bq[1][kk], aS1);
    }

    // ---- register softmax partial max (v-reduce + half-wave shfl) ----
    {
      float pm0 = max8(aS0), pm1 = max8(aS1);
      pm0 = fmaxf(pm0, __shfl_xor(pm0, 16, 32));
      pm1 = fmaxf(pm1, __shfl_xor(pm1, 16, 32));
      if (lane < 16) {
        red1[wave >> 1][sm0 + lane]      = pm0;
        red1[wave >> 1][sm0 + 16 + lane] = pm1;
      }
    }
    __syncthreads();                               // bar1
    if (t < 64) {
      float bm = fmaxf(fmaxf(red1[0][t], red1[1][t]),
                       fmaxf(red1[2][t], red1[3][t]));
      float Mo = Mld[t], Mn = fmaxf(Mo, bm);
      scld[t] = __expf(Mo - Mn);
      Mld[t]  = Mn;
    }
    __syncthreads();                               // bar2

    // ---- P = exp(S - M): packed b128 store per column; partial sums ----
    {
      const float Mn0 = Mld[sm0 + col];
      const float Mn1 = Mld[sm0 + 16 + col];
      v8f e0, e1;
      float ps0 = 0.f, ps1 = 0.f;
      #pragma unroll
      for (int v = 0; v < 8; ++v) {
        float a0 = __expf(aS0[v] - Mn0);
        float a1 = __expf(aS1[v] - Mn1);
        e0[v] = a0; e1[v] = a1;
        ps0 += a0; ps1 += a1;
      }
      const int nbase = sn0 + hsel;                // lane's 8 rows, contiguous n
      *reinterpret_cast<uint4*>(&PT[(sm0 + col) * TS + nbase])      = packD(e0);
      *reinterpret_cast<uint4*>(&PT[(sm0 + 16 + col) * TS + nbase]) = packD(e1);
      ps0 += __shfl_xor(ps0, 16, 32);
      ps1 += __shfl_xor(ps1, 16, 32);
      if (lane < 16) {
        red2[wave >> 1][sm0 + lane]      = ps0;
        red2[wave >> 1][sm0 + 16 + lane] = ps1;
      }
    }
    __syncthreads();                               // bar3: PT + sums visible
    if (t < 64)
      Lld[t] = Lld[t] * scld[t] + (red2[0][t] + red2[1][t] + red2[2][t] + red2[3][t]);

    // ---- rescale O^T (per-v scale vector) ----
    #pragma unroll
    for (int mf = 0; mf < 4; ++mf) {
      const float4 s0 = *reinterpret_cast<const float4*>(&scld[mf * 16 + hsel]);
      const float4 s1 = *reinterpret_cast<const float4*>(&scld[mf * 16 + hsel + 4]);
      #pragma unroll
      for (int cf = 0; cf < 4; ++cf) {
        accO[mf][cf][0] *= s0.x; accO[mf][cf][1] *= s0.y;
        accO[mf][cf][2] *= s0.z; accO[mf][cf][3] *= s0.w;
        accO[mf][cf][4] *= s1.x; accO[mf][cf][5] *= s1.y;
        accO[mf][cf][6] *= s1.z; accO[mf][cf][7] *= s1.w;
      }
    }
    // ---- O^T += P^T(m,n) . hv^T(n,c):  A = PT rows(m), B = hv via BT path ----
    #pragma unroll
    for (int kk = 0; kk < 2; ++kk) {
      v16bf ap[4];
      #pragma unroll
      for (int mf = 0; mf < 4; ++mf)
        ap[mf] = fragA(PT, TS, mf * 16, kk * 32, lane);
      #pragma unroll
      for (int cf = 0; cf < 4; ++cf) {
        v16bf bh = fragBT(hb, NPIX, cbase + cf * 16, nblk + kk * 32, lane);
        #pragma unroll
        for (int mf = 0; mf < 4; ++mf)
          accO[mf][cf] = wmma_bf16(ap[mf], bh, accO[mf][cf]);
      }
    }
    // pipelining: bar1..bar3 of the next iteration protect PT/scld/red reuse
  }

  // ---- finalize: out[c,m] = gamma * O^T[m,c] / L[m] + x[c,m] (vectorized) ----
  __syncthreads();
  if (t < 64) scld[t] = 1.0f / Lld[t];
  __syncthreads();

  const float gam = gamma[0];
  const float* xb = x   + (size_t)b * C * NPIX;
  float*       ob = out + (size_t)b * C * NPIX;
  #pragma unroll
  for (int mf = 0; mf < 4; ++mf) {
    const int msel = mf * 16 + hsel;
    const float4 i0 = *reinterpret_cast<const float4*>(&scld[msel]);
    const float4 i1 = *reinterpret_cast<const float4*>(&scld[msel + 4]);
    #pragma unroll
    for (int cf = 0; cf < 4; ++cf) {
      const int c = cbase + cf * 16 + col;
      const size_t base = (size_t)c * NPIX + m0 + msel;
      float4 x0 = *reinterpret_cast<const float4*>(&xb[base]);
      float4 x1 = *reinterpret_cast<const float4*>(&xb[base + 4]);
      float4 o0, o1;
      o0.x = gam * accO[mf][cf][0] * i0.x + x0.x;
      o0.y = gam * accO[mf][cf][1] * i0.y + x0.y;
      o0.z = gam * accO[mf][cf][2] * i0.z + x0.z;
      o0.w = gam * accO[mf][cf][3] * i0.w + x0.w;
      o1.x = gam * accO[mf][cf][4] * i1.x + x1.x;
      o1.y = gam * accO[mf][cf][5] * i1.y + x1.y;
      o1.z = gam * accO[mf][cf][6] * i1.z + x1.z;
      o1.w = gam * accO[mf][cf][7] * i1.w + x1.w;
      *reinterpret_cast<float4*>(&ob[base])     = o0;
      *reinterpret_cast<float4*>(&ob[base + 4]) = o1;
    }
  }
}

// =====================================================================
extern "C" void kernel_launch(void* const* d_in, const int* in_sizes, int n_in,
                              void* d_out, int out_size, void* d_ws, size_t ws_size,
                              hipStream_t stream) {
  (void)in_sizes; (void)n_in; (void)out_size; (void)ws_size;
  const float* x     = (const float*)d_in[0];
  const float* Wq    = (const float*)d_in[1];
  const float* Wk    = (const float*)d_in[2];
  const float* Wv    = (const float*)d_in[3];
  const float* gamma = (const float*)d_in[4];
  float* outp = (float*)d_out;

  // Workspace: bf16 fT|gT|hv = 20,971,520 bytes
  unsigned short* fgh = (unsigned short*)d_ws;

  dim3 gProj(NPIX / 64, ROWS / 64, BATCH);   // (64, 10, 4)
  proj_kernel<<<gProj, 256, 0, stream>>>(x, Wq, Wk, Wv, fgh);

  dim3 gAttn(NPIX / 64, BATCH);              // (64, 4)
  attn_kernel<<<gAttn, 256, 0, stream>>>(x, fgh, gamma, outp);
}
